// FasterRCNNPredictor_42494406427388
// MI455X (gfx1250) — compile-verified, matrix operand order checked by reading, not run on hardware
//
#include <hip/hip_runtime.h>
#include <hip/hip_bf16.h>
#include <stdint.h>
#include <math.h>

// Problem constants (from reference)
#define BATCH 8
#define NBOX  2000
#define NCLS  81
#define NPAD  2048          // power-of-two padding for sort / masks
#define IOU_THRESH 0.5f
#define CLS_OFFSET 100000.0f

// d_out float offsets (outputs concatenated flat, in return order)
#define OUT0 ((size_t)0)          // nms_reg      B*N*4  = 64000
#define OUT1 ((size_t)64000)      // nms_cls      B*N*2  = 32000
#define OUT2 ((size_t)96000)      // rcnn_reg_adj B*N*4  = 64000
#define OUT3 ((size_t)160000)     // probs        B*N*81 = 1296000
#define OUT4 ((size_t)1456000)    // reg_out      B*N*4  = 64000
#define OUT5 ((size_t)1520000)    // cls_out      B*N*81 = 1296000
#define OUT6 ((size_t)2816000)    // keep         B*N    = 16000

typedef __attribute__((ext_vector_type(16))) _Float16 v16h;
typedef __attribute__((ext_vector_type(8)))  float    v8f;
typedef __attribute__((ext_vector_type(4)))  int      v4i;

// CDNA5 async global->LDS copy path (ASYNCcnt-tracked), guarded so the host
// pass and older toolchains fall back to plain loads.
#if defined(__HIP_DEVICE_COMPILE__) && \
    __has_builtin(__builtin_amdgcn_global_load_async_to_lds_b128) && \
    __has_builtin(__builtin_amdgcn_s_wait_asynccnt)
#define HAS_ASYNC_LDS 1
#else
#define HAS_ASYNC_LDS 0
#endif

// ---------------------------------------------------------------------------
// Kernel A: rounding + reg adjust + softmax (WMMA row-sum) + argmax/score
// 128 threads = 4 waves; each wave owns 16 rows. Grid = B * (NPAD/64).
// ---------------------------------------------------------------------------
__global__ __launch_bounds__(128)
void frcnn_softmax_kernel(const float* __restrict__ nms_reg,
                          const float* __restrict__ nms_cls,
                          const float* __restrict__ rcnn_reg,
                          const float* __restrict__ rcnn_cls,
                          const int*   __restrict__ red_p,
                          float* __restrict__ out,
                          float* __restrict__ scores,
                          int*   __restrict__ clsarg,
                          int*   __restrict__ keepbuf)
{
    __shared__ float buf[4][16][96];   // exp workspace, padded 81 -> 96
    __shared__ float maxv[4][16];
    __shared__ float sums[4][16];
    __shared__ int   amax[4][16];

    const int w    = threadIdx.x >> 5;
    const int lane = threadIdx.x & 31;
    const int b    = blockIdx.x >> 5;            // image
    const int base = (blockIdx.x & 31) * 64 + w * 16;  // first row of this wave

    // Stage logits into LDS (zeros for pad cols / pad rows)
    for (int idx = lane; idx < 16 * 96; idx += 32) {
        int r = idx / 96, c = idx % 96;
        int n = base + r;
        float v = 0.0f;
        if (c < NCLS && n < NBOX)
            v = rcnn_cls[((size_t)(b * NBOX + n)) * NCLS + c];
        buf[w][r][c] = (c < NCLS) ? v : 0.0f;
    }
    __syncthreads();

    // Per-row max + argmax (first-max tie break, matches jnp.argmax)
    if (lane < 16) {
        float m = -3.402823466e38f; int a = 0;
        for (int c = 0; c < NCLS; ++c) {
            float v = buf[w][lane][c];
            if (v > m) { m = v; a = c; }
        }
        maxv[w][lane] = m; amax[w][lane] = a;
    }
    __syncthreads();

    // exp(x - max); pad cols -> 0 so they contribute nothing to the sum
    for (int idx = lane; idx < 16 * 96; idx += 32) {
        int r = idx / 96, c = idx % 96;
        buf[w][r][c] = (c < NCLS) ? __expf(buf[w][r][c] - maxv[w][r]) : 0.0f;
    }
    __syncthreads();

    // --- WMMA row-sum: D = A(16x32 f16) * ones(32x16) accumulated over 3 K-chunks.
    // A fragment layout (ISA 7.12.2, 16-bit A 16x32): lane<16 owns K 0-7 / 16-23,
    // lane>=16 owns K 8-15 / 24-31 of row (lane%16). All-ones B makes the sum
    // exact under any K permutation.
    const int row  = lane & 15;
    const int koff = (lane < 16) ? 0 : 8;
    v16h ones;
    for (int t = 0; t < 16; ++t) ones[t] = (_Float16)1.0f;
    v8f acc = {};
    for (int k0 = 0; k0 < 96; k0 += 32) {
        v16h a;
        for (int t = 0; t < 8; ++t) {
            a[t]     = (_Float16)buf[w][row][k0 + koff + t];
            a[8 + t] = (_Float16)buf[w][row][k0 + koff + 16 + t];
        }
        acc = __builtin_amdgcn_wmma_f32_16x16x32_f16(
                  false, a, false, ones, (short)0, acc, false, false);
    }
    // C/D layout: lane 0 VGPR r = (M=r, N=0); lane 16 VGPR r = (M=8+r, N=0)
    if (lane == 0)  for (int r2 = 0; r2 < 8; ++r2) sums[w][r2]     = acc[r2];
    if (lane == 16) for (int r2 = 0; r2 < 8; ++r2) sums[w][8 + r2] = acc[r2];
    __syncthreads();

    // probs out
    for (int idx = lane; idx < 16 * NCLS; idx += 32) {
        int r = idx / NCLS, c = idx % NCLS;
        int n = base + r;
        if (n < NBOX)
            out[OUT3 + ((size_t)(b * NBOX + n)) * NCLS + c] = buf[w][r][c] / sums[w][r];
    }

    // Per-row scalars + box math (16 lanes, one row each)
    if (lane < 16) {
        int n  = base + lane;
        int gi = b * NPAD + n;
        if (n < NBOX) {
            // max prob = exp(max-max)/sum = 1/sum
            scores[gi]  = 1.0f / sums[w][lane];
            clsarg[gi]  = amax[w][lane];
            keepbuf[gi] = 0;

            float red = (float)red_p[0];
            size_t r4 = ((size_t)(b * NBOX + n)) * 4;
            float t0 = nms_reg[r4 + 0], l0 = nms_reg[r4 + 1];
            float b0 = nms_reg[r4 + 2], r0 = nms_reg[r4 + 3];
            float rt = floorf(t0 * red) / red, rl = floorf(l0 * red) / red;
            float rb = ceilf(b0 * red) / red,  rr = ceilf(r0 * red) / red;
            out[OUT2 + r4 + 0] = rcnn_reg[r4 + 0] + rt;
            out[OUT2 + r4 + 1] = rcnn_reg[r4 + 1] + rl;
            out[OUT2 + r4 + 2] = rcnn_reg[r4 + 2] + rb;
            out[OUT2 + r4 + 3] = rcnn_reg[r4 + 3] + rr;
            // passthrough copies
            out[OUT0 + r4 + 0] = t0; out[OUT0 + r4 + 1] = l0;
            out[OUT0 + r4 + 2] = b0; out[OUT0 + r4 + 3] = r0;
            size_t r2i = ((size_t)(b * NBOX + n)) * 2;
            out[OUT1 + r2i + 0] = nms_cls[r2i + 0];
            out[OUT1 + r2i + 1] = nms_cls[r2i + 1];
        } else {
            scores[gi]  = -__builtin_inff();
            clsarg[gi]  = 0;
            keepbuf[gi] = 0;
        }
    }
}

// ---------------------------------------------------------------------------
// Kernel B: per-image bitonic sort of 2048 entries by (score desc, idx asc)
// (reproduces stable argsort of -masked_score), then gather offset boxes.
// ---------------------------------------------------------------------------
__global__ __launch_bounds__(1024)
void frcnn_sort_kernel(const float* __restrict__ scores,
                       const int*   __restrict__ clsarg,
                       const float* __restrict__ reg_adj,   // d_out + OUT2
                       float* __restrict__ sbox,
                       int*   __restrict__ svalid,
                       int*   __restrict__ order)
{
    __shared__ float key[NPAD];
    __shared__ int   sid[NPAD];
    const int b   = blockIdx.x;
    const int tid = threadIdx.x;

    for (int i = tid; i < NPAD; i += 1024) {
        int gi = b * NPAD + i;
        key[i] = (clsarg[gi] != 0) ? scores[gi] : -__builtin_inff();
        sid[i] = i;
    }
    __syncthreads();

    for (int k = 2; k <= NPAD; k <<= 1) {
        for (int j = k >> 1; j > 0; j >>= 1) {
            for (int i = tid; i < NPAD; i += 1024) {
                int ixj = i ^ j;
                if (ixj > i) {
                    bool dir = ((i & k) == 0);           // descending segment
                    float ka = key[i], kb = key[ixj];
                    int   ia = sid[i], ib = sid[ixj];
                    bool g = (kb > ka) || (kb == ka && ib < ia); // ixj "greater"
                    if (g == dir) {
                        key[i] = kb; key[ixj] = ka;
                        sid[i] = ib; sid[ixj] = ia;
                    }
                }
            }
            __syncthreads();
        }
    }

    for (int i = tid; i < NPAD; i += 1024) {
        int o  = sid[i];
        int gi = b * NPAD + i;
        order[gi] = o;
        int vflag = (key[i] != -__builtin_inff()) ? 1 : 0;
        svalid[gi] = vflag;
        float bx0 = 0.f, bx1 = 0.f, bx2 = 0.f, bx3 = 0.f;
        if (vflag) {  // valid => o < NBOX
            float off = (float)clsarg[b * NPAD + o] * CLS_OFFSET;
            size_t r4 = ((size_t)(b * NBOX + o)) * 4;
            bx0 = reg_adj[r4 + 0] + off; bx1 = reg_adj[r4 + 1] + off;
            bx2 = reg_adj[r4 + 2] + off; bx3 = reg_adj[r4 + 3] + off;
        }
        size_t s4 = (size_t)gi * 4;
        sbox[s4 + 0] = bx0; sbox[s4 + 1] = bx1;
        sbox[s4 + 2] = bx2; sbox[s4 + 3] = bx3;
    }
}

// ---------------------------------------------------------------------------
// Kernel C: suppression bitmask build. Block = 256 rows vs one 64-col block.
// Column tile staged via async global->LDS (ASYNCcnt) when available.
// Grid = B * 32 colblocks * 8 rowchunks = 2048 blocks.
// ---------------------------------------------------------------------------
__global__ __launch_bounds__(256)
void frcnn_mask_kernel(const float* __restrict__ sbox,
                       unsigned long long* __restrict__ masks)
{
    __shared__ float4 cbv[64];
    const int blk = blockIdx.x;
    const int b   = blk / 256;
    const int rem = blk % 256;
    const int cbi = rem / 8;
    const int rc  = rem % 8;
    const int row = rc * 256 + threadIdx.x;
    const float4* __restrict__ sbox4 = (const float4*)sbox;

    if (threadIdx.x < 64) {
        int src = b * NPAD + cbi * 64 + threadIdx.x;
#if HAS_ASYNC_LDS
        __builtin_amdgcn_global_load_async_to_lds_b128(
            (__attribute__((address_space(1))) v4i*)(v4i*)(void*)&sbox4[src],
            (__attribute__((address_space(3))) v4i*)(v4i*)(void*)&cbv[threadIdx.x],
            0, 0);
#else
        cbv[threadIdx.x] = sbox4[src];
#endif
    }
#if HAS_ASYNC_LDS
    __builtin_amdgcn_s_wait_asynccnt(0);   // issuing waves drain ASYNCcnt
#endif
    __syncthreads();                       // LDS tile visible block-wide

    float4 rbx = sbox4[b * NPAD + row];    // global_load_b128
    float areaR = (rbx.z - rbx.x) * (rbx.w - rbx.y);

    unsigned long long m = 0ull;
    for (int jj = 0; jj < 64; ++jj) {
        int j = cbi * 64 + jj;
        if (j > row) {
            float4 cj = cbv[jj];
            float it = fmaxf(rbx.x, cj.x), il = fmaxf(rbx.y, cj.y);
            float ib = fminf(rbx.z, cj.z), ir = fminf(rbx.w, cj.w);
            float inter = fmaxf(ib - it, 0.0f) * fmaxf(ir - il, 0.0f);
            float uni   = areaR + (cj.z - cj.x) * (cj.w - cj.y) - inter;
            float iou   = inter / fmaxf(uni, 1e-9f);
            if (iou > IOU_THRESH) m |= (1ull << jj);
        }
    }
    masks[((size_t)(b * NPAD) + row) * 32 + cbi] = m;
}

// ---------------------------------------------------------------------------
// Kernel D: serial greedy scan, one wave32 per image. The 2048-bit removal
// state lives in 32 x 64-bit lane registers; broadcast via __shfl.
// global_prefetch hides L2 latency on the dependent-load chain.
// ---------------------------------------------------------------------------
__global__ __launch_bounds__(32)
void frcnn_scan_kernel(const unsigned long long* __restrict__ masks,
                       const int* __restrict__ svalid,
                       const int* __restrict__ order,
                       int* __restrict__ keepbuf)
{
    const int b    = blockIdx.x;
    const int lane = threadIdx.x;
    unsigned long long rem = 0ull;

    for (int i = 0; i < NPAD; ++i) {
        if (i + 4 < NPAD)   // prefetch mask row i+4 (emits global_prefetch_b8)
            __builtin_prefetch(&masks[((size_t)(b * NPAD) + i + 4) * 32 + lane], 0, 1);

        int wsel = i >> 6, bit = i & 63;
        int lo = __shfl((int)(unsigned)(rem & 0xffffffffull), wsel, 32);
        int hi = __shfl((int)(unsigned)(rem >> 32), wsel, 32);
        unsigned long long rw =
            ((unsigned long long)(unsigned)hi << 32) | (unsigned)(unsigned int)lo;
        bool kept = (((rw >> bit) & 1ull) == 0ull) && (svalid[b * NPAD + i] != 0);
        if (kept) {
            rem |= masks[((size_t)(b * NPAD) + i) * 32 + lane];
            if (lane == 0) {
                int o = order[b * NPAD + i];   // valid => o < NBOX
                keepbuf[b * NPAD + o] = 1;
            }
        }
    }
}

// ---------------------------------------------------------------------------
// Kernel E: masked outputs + keep. One block per (b,n), 96 threads.
// ---------------------------------------------------------------------------
__global__ __launch_bounds__(96)
void frcnn_finalize_kernel(const int* __restrict__ keepbuf,
                           float* __restrict__ out)
{
    const int b = blockIdx.x / NBOX;
    const int n = blockIdx.x % NBOX;
    const int kp = keepbuf[b * NPAD + n];
    const int t = threadIdx.x;
    size_t rowc = ((size_t)(b * NBOX + n)) * NCLS;
    size_t row4 = ((size_t)(b * NBOX + n)) * 4;
    if (t < NCLS) {
        out[OUT5 + rowc + t] = kp ? out[OUT3 + rowc + t] : 0.0f;
    } else if (t < 85) {
        out[OUT4 + row4 + (t - 81)] = kp ? out[OUT2 + row4 + (t - 81)] : 0.0f;
    } else if (t == 85) {
        out[OUT6 + (size_t)(b * NBOX + n)] = kp ? 1.0f : 0.0f;
    }
}

// ---------------------------------------------------------------------------
extern "C" void kernel_launch(void* const* d_in, const int* in_sizes, int n_in,
                              void* d_out, int out_size, void* d_ws, size_t ws_size,
                              hipStream_t stream) {
    (void)in_sizes; (void)n_in; (void)out_size; (void)ws_size;

    const float* nms_reg  = (const float*)d_in[0];
    const float* nms_cls  = (const float*)d_in[1];
    const float* rcnn_reg = (const float*)d_in[2];
    const float* rcnn_cls = (const float*)d_in[3];
    const int*   red      = (const int*)d_in[4];
    float* out = (float*)d_out;

    // workspace layout (bytes) — fully rewritten every call
    char* ws = (char*)d_ws;
    float* scores  = (float*)(ws + 0);                      // B*2048 f32  (64KB)
    int*   clsarg  = (int*)  (ws + 65536);                  // B*2048 i32  (64KB)
    int*   order   = (int*)  (ws + 131072);                 // B*2048 i32  (64KB)
    float* sbox    = (float*)(ws + 196608);                 // B*2048*4 f32 (256KB)
    int*   svalid  = (int*)  (ws + 458752);                 // B*2048 i32  (64KB)
    unsigned long long* masks = (unsigned long long*)(ws + 524288); // B*2048*32 u64 (4MB)
    int*   keepbuf = (int*)  (ws + 524288 + 4194304);       // B*2048 i32  (64KB)

    frcnn_softmax_kernel<<<dim3(BATCH * (NPAD / 64)), dim3(128), 0, stream>>>(
        nms_reg, nms_cls, rcnn_reg, rcnn_cls, red, out, scores, clsarg, keepbuf);

    frcnn_sort_kernel<<<dim3(BATCH), dim3(1024), 0, stream>>>(
        scores, clsarg, out + OUT2, sbox, svalid, order);

    frcnn_mask_kernel<<<dim3(BATCH * 32 * 8), dim3(256), 0, stream>>>(sbox, masks);

    frcnn_scan_kernel<<<dim3(BATCH), dim3(32), 0, stream>>>(masks, svalid, order, keepbuf);

    frcnn_finalize_kernel<<<dim3(BATCH * NBOX), dim3(96), 0, stream>>>(keepbuf, out);
}
